// CrystalGraphConvNet_64115271795213
// MI455X (gfx1250) — matrix operand order repaired and checked by reading, not compiled, not run on hardware
//
#include <hip/hip_runtime.h>
#include <hip/hip_bf16.h>

typedef __bf16 bf16;
typedef __attribute__((ext_vector_type(16))) __bf16 v16bf;
typedef __attribute__((ext_vector_type(8)))  __bf16 v8bf;
typedef __attribute__((ext_vector_type(8)))  float  v8f;

#define NATOMS 20000
#define MNBR   12
#define NBRF   64
#define AF     256
#define ORIGF  92
#define KDIM   576   // 2*AF + NBRF
#define N1     512   // 2*AF
#define NCONV  3
#define BCRYS  200
#define KCRYS  50

__device__ __forceinline__ float softplus_f(float v) {
    // numerically stable log(1+exp(v))
    return fmaxf(v, 0.f) + log1pf(__expf(-fabsf(v)));
}

// A-fragment (16x32 bf16): lane holds row (lane&15); lanes>=16 shifted +8 K.
// K chunks: [+0B..15B] and [+32B..47B] (lane<16), +16B bump for lane>=16.
__device__ __forceinline__ v16bf frag_a(const bf16* p, int lane) {
    const char* c = (const char*)p + ((lane & 16) ? 16 : 0);
    v8bf lo = *(const v8bf*)(c);
    v8bf hi = *(const v8bf*)(c + 32);
    return __builtin_shufflevector(lo, hi, 0,1,2,3,4,5,6,7,8,9,10,11,12,13,14,15);
}
// B-fragment (32x16 bf16): lane holds column (lane&15); lane<16 => K0..7,16..23,
// lane>=16 => K8..15,24..31 (same order as A).
__device__ __forceinline__ v16bf frag_b(const bf16* p, int lane) {
    const char* c = (const char*)p + ((lane & 16) ? 16 : 0);
    v8bf lo = *(const v8bf*)(c);
    v8bf hi = *(const v8bf*)(c + 32);
    return __builtin_shufflevector(lo, hi, 0,1,2,3,4,5,6,7,8,9,10,11,12,13,14,15);
}

// ---------------------------------------------------------------- prep ----
__global__ __launch_bounds__(256) void prep_weights(
    const float* __restrict__ w1, const float* __restrict__ w2,
    bf16* __restrict__ w1t, bf16* __restrict__ w2t, float* __restrict__ bnacc)
{
    int i = blockIdx.x * 256 + threadIdx.x;
    const int tot1 = NCONV * KDIM * N1;      // 884736
    const int tot2 = NCONV * N1 * AF;        // 393216
    if (i < tot1) {
        int l = i / (KDIM * N1);
        int r = i % (KDIM * N1);
        int k = r / N1, n = r % N1;
        w1t[((size_t)l * N1 + n) * KDIM + k] = (bf16)w1[i];
    } else if (i < tot1 + tot2) {
        int j = i - tot1;
        int l = j / (N1 * AF);
        int r = j % (N1 * AF);
        int k = r / AF, n = r % AF;
        w2t[((size_t)l * AF + n) * N1 + k] = (bf16)w2[j];
    }
    if (i < NCONV * 512) bnacc[i] = 0.f;
}

__global__ __launch_bounds__(256) void prep_nbr(
    const float* __restrict__ nbr, bf16* __restrict__ nbr_bf, int n)
{
    int i = blockIdx.x * 256 + threadIdx.x;
    if (i < n) nbr_bf[i] = (bf16)nbr[i];
}

// ------------------------------------------------------------- embed -----
__global__ __launch_bounds__(256) void embed_k(
    const float* __restrict__ atom_fea, const float* __restrict__ emb_w,
    const float* __restrict__ emb_b, float* __restrict__ x, bf16* __restrict__ x_bf)
{
    __shared__ float row[ORIGF];
    int n = blockIdx.x, t = threadIdx.x;
    if (t < ORIGF) row[t] = atom_fea[n * ORIGF + t];
    __syncthreads();
    float acc = emb_b[t];
    for (int k = 0; k < ORIGF; ++k) acc = fmaf(row[k], emb_w[k * AF + t], acc);
    float o = softplus_f(acc);
    x[(size_t)n * AF + t]    = o;
    x_bf[(size_t)n * AF + t] = (bf16)o;
}

// --------------------------------------------------- conv layer GEMM -----
// Block: 4 atoms -> 48 rows (12 per atom: j=0..5 relu branch, j=6..11 softplus),
// 256 output cols (blockIdx.y selects half of N1=512), K=576.
#define APB 4
__global__ __launch_bounds__(256) void conv_gemm(
    const bf16* __restrict__ x_bf, const bf16* __restrict__ nbr_bf,
    const int* __restrict__ nbr_idx, const bf16* __restrict__ w1t,
    const float* __restrict__ b1, bf16* __restrict__ ns_bf, int layer)
{
    __shared__ int  sidx[APB * MNBR];
    __shared__ uint smem[13440];           // A(8320w) + B(5120w), union epilogue

    int tid = threadIdx.x, lane = tid & 31, wave = tid >> 5;
    int atom0 = blockIdx.x * APB;
    int colHalf = blockIdx.y;

    if (tid < APB * MNBR)
        sidx[tid] = nbr_idx[(atom0 + tid / MNBR) * MNBR + (tid % MNBR)];
    __syncthreads();

    // --- stage compact A: per atom 2048 uint words (own 128 | 12 nbr x128 | 12 nf x32)
    const uint* xw = (const uint*)x_bf;
    const uint* nw = (const uint*)nbr_bf;
    for (int i = tid; i < APB * 2048; i += 256) {
        int a = i >> 11, t = i & 2047;
        uint val;
        if (t < 128) {
            val = xw[(size_t)(atom0 + a) * 128 + t];
        } else if (t < 1664) {
            int j = (t - 128) >> 7, tt = (t - 128) & 127;
            val = xw[(size_t)sidx[a * MNBR + j] * 128 + tt];
        } else {
            int t2 = t - 1664;
            int j = t2 >> 5, tt = t2 & 31;
            val = nw[((size_t)(atom0 + a) * MNBR + j) * 32 + tt];
        }
        smem[a * 2080 + t] = val;
    }

    int wcol = wave * 32;
    v8f acc[3][2] = {};
    const uint* bw = (const uint*)(w1t + (size_t)layer * N1 * KDIM);

    for (int kc = 0; kc < KDIM / 32; ++kc) {
        int kb = kc * 32;
        __syncthreads();
        // stage B chunk: 256 n-rows x 32 bf16 (stride 40 elems = 20 words)
        for (int i = tid; i < 4096; i += 256) {
            int n = i >> 4, t = i & 15;
            smem[8320 + n * 20 + t] = bw[((size_t)(colHalf * 256 + n) * (KDIM / 2)) + (kb >> 1) + t];
        }
        __syncthreads();

        const bf16* sB = (const bf16*)(smem + 8320);
        v16bf bf0 = frag_b(sB + (wcol + 0  + (lane & 15)) * 40, lane);
        v16bf bf1 = frag_b(sB + (wcol + 16 + (lane & 15)) * 40, lane);

        for (int rt = 0; rt < 3; ++rt) {
            int r = rt * 16 + (lane & 15);
            int a = r / 12, j = r % 12;
            int off;
            if (kb < 256)      off = a * 4160 + kb;
            else if (kb < 512) off = a * 4160 + 256 + j * 256 + (kb - 256);
            else               off = a * 4160 + 3328 + j * 64 + (kb - 512);
            v16bf af = frag_a((const bf16*)smem + off, lane);
            acc[rt][0] = __builtin_amdgcn_wmma_f32_16x16x32_bf16(
                false, af, false, bf0, (short)0, acc[rt][0], false, false);
            acc[rt][1] = __builtin_amdgcn_wmma_f32_16x16x32_bf16(
                false, af, false, bf1, (short)0, acc[rt][1], false, false);
        }
    }
    __syncthreads();

    // --- epilogue: bias + branch activation, stage into LDS (union), reduce over j
    float* sOut = (float*)smem;            // 48 rows x stride 260 f32
    for (int rt = 0; rt < 3; ++rt)
        for (int ct = 0; ct < 2; ++ct) {
            int n_local = wcol + ct * 16 + (lane & 15);
            float bias = b1[layer * N1 + colHalf * 256 + n_local];
            for (int e = 0; e < 8; ++e) {
                int Mrow = e + ((lane & 16) ? 8 : 0);
                int row = rt * 16 + Mrow;
                float v = acc[rt][ct][e] + bias;
                float g = ((row % 12) < 6) ? fmaxf(v, 0.f) : softplus_f(v);
                sOut[row * 260 + n_local] = g;
            }
        }
    __syncthreads();
    for (int i = tid; i < APB * 256; i += 256) {
        int a = i >> 8, c = i & 255;
        float s = 0.f;
        for (int r = 0; r < 12; ++r) s += sOut[(a * 12 + r) * 260 + c];
        ns_bf[(size_t)(atom0 + a) * N1 + colHalf * 256 + c] = (bf16)s;
    }
}

// ---------------------------------------- ns @ w2 + b2, fused BN stats ---
#define MT2 64
__global__ __launch_bounds__(256) void gemm2_stats(
    const bf16* __restrict__ ns, const bf16* __restrict__ w2t,
    const float* __restrict__ b2, float* __restrict__ y,
    float* __restrict__ bnacc, int layer)
{
    __shared__ bf16 sA[MT2 * 40];
    __shared__ bf16 sB[AF * 40];
    __shared__ float sSum[AF], sSq[AF];

    int tid = threadIdx.x, lane = tid & 31, wave = tid >> 5;
    int row0 = blockIdx.x * MT2;
    sSum[tid] = 0.f; sSq[tid] = 0.f;

    int rt = wave & 3;
    int colb = (wave >> 2) * 128;
    v8f acc[8] = {};
    const uint* aw  = (const uint*)ns;
    const uint* bwp = (const uint*)(w2t + (size_t)layer * AF * N1);
    uint* sAw = (uint*)sA; uint* sBw = (uint*)sB;

    for (int kc = 0; kc < N1 / 32; ++kc) {
        int kb = kc * 32;
        __syncthreads();
        for (int i = tid; i < 1024; i += 256) {     // A: 64 rows x 16 words
            int r = i >> 4, t = i & 15;
            int row = row0 + r;
            sAw[r * 20 + t] = (row < NATOMS) ? aw[(size_t)row * (N1 / 2) + (kb >> 1) + t] : 0u;
        }
        for (int i = tid; i < 4096; i += 256) {     // B: 256 cols x 16 words
            int n = i >> 4, t = i & 15;
            sBw[n * 20 + t] = bwp[(size_t)n * (N1 / 2) + (kb >> 1) + t];
        }
        __syncthreads();
        v16bf af = frag_a(sA + (rt * 16 + (lane & 15)) * 40, lane);
        for (int ct = 0; ct < 8; ++ct) {
            v16bf bfr = frag_b(sB + (colb + ct * 16 + (lane & 15)) * 40, lane);
            acc[ct] = __builtin_amdgcn_wmma_f32_16x16x32_bf16(
                false, af, false, bfr, (short)0, acc[ct], false, false);
        }
    }
    __syncthreads();

    for (int ct = 0; ct < 8; ++ct) {
        int n = colb + ct * 16 + (lane & 15);
        float bias = b2[layer * AF + n];
        float s1 = 0.f, s2 = 0.f;
        for (int e = 0; e < 8; ++e) {
            int row = row0 + rt * 16 + e + ((lane & 16) ? 8 : 0);
            if (row < NATOMS) {
                float v = acc[ct][e] + bias;
                y[(size_t)row * AF + n] = v;
                s1 += v; s2 += v * v;
            }
        }
        s1 += __shfl_xor(s1, 16, 32);
        s2 += __shfl_xor(s2, 16, 32);
        if (lane < 16) { atomicAdd(&sSum[n], s1); atomicAdd(&sSq[n], s2); }
    }
    __syncthreads();
    atomicAdd(&bnacc[layer * 512 + tid % 256 + (tid < 256 ? 0 : 256)],
              (tid < 256) ? sSum[tid] : sSq[tid - 256]);
}

// --------------------------------------------- BN apply + residual -------
__global__ __launch_bounds__(256) void bn_apply(
    float* __restrict__ x, const float* __restrict__ y,
    const float* __restrict__ bnacc, const float* __restrict__ gamma,
    const float* __restrict__ beta, bf16* __restrict__ x_bf, int layer)
{
    int i = blockIdx.x * 256 + threadIdx.x;     // 20000*256 total
    int col = i & (AF - 1);
    float mu = bnacc[layer * 512 + col] * (1.f / NATOMS);
    float m2 = bnacc[layer * 512 + 256 + col] * (1.f / NATOMS);
    float inv = rsqrtf(m2 - mu * mu + 1e-5f);
    float v = (y[i] - mu) * inv * gamma[layer * AF + col] + beta[layer * AF + col];
    float o = softplus_f(x[i] + v);
    x[i] = o;
    x_bf[i] = (bf16)o;
}

// ------------------------------------------------------ pooling + head ---
__global__ __launch_bounds__(256) void head_k(
    const float* __restrict__ x, const int* __restrict__ m1_idx,
    const float* __restrict__ m2, const float* __restrict__ fc_w,
    const float* __restrict__ fc_b, const float* __restrict__ out_w,
    const float* __restrict__ out_b, float* __restrict__ out)
{
    __shared__ float sp[AF + 4];
    __shared__ float red[256];
    int b = blockIdx.x, t = threadIdx.x;
    float s = 0.f;
    for (int k = 0; k < KCRYS; ++k) {
        int idx = m1_idx[b * KCRYS + k];
        s += x[(size_t)idx * AF + t];
    }
    sp[t] = softplus_f(s * (1.f / KCRYS));
    if (t < 4) sp[AF + t] = softplus_f(m2[b * 4 + t]);
    __syncthreads();
    float h = fc_b[t];
    for (int k = 0; k < AF + 4; ++k) h = fmaf(sp[k], fc_w[k * 256 + t], h);
    h = softplus_f(h);
    red[t] = h * out_w[t];
    __syncthreads();
    for (int st = 128; st > 0; st >>= 1) {
        if (t < st) red[t] += red[t + st];
        __syncthreads();
    }
    if (t == 0) out[b] = red[0] + out_b[0];
}

// --------------------------------------------------------------- launch --
extern "C" void kernel_launch(void* const* d_in, const int* in_sizes, int n_in,
                              void* d_out, int out_size, void* d_ws, size_t ws_size,
                              hipStream_t stream)
{
    const float* atom_fea = (const float*)d_in[0];
    const float* nbr_fea  = (const float*)d_in[1];
    const int*   nbr_idx  = (const int*)d_in[2];
    const int*   m1_index = (const int*)d_in[3];
    const float* m2_fea   = (const float*)d_in[5];
    const float* emb_w    = (const float*)d_in[6];
    const float* emb_b    = (const float*)d_in[7];
    const float* w1       = (const float*)d_in[8];
    const float* b1       = (const float*)d_in[9];
    const float* w2       = (const float*)d_in[10];
    const float* b2       = (const float*)d_in[11];
    const float* gamma    = (const float*)d_in[12];
    const float* beta     = (const float*)d_in[13];
    const float* fc_w     = (const float*)d_in[14];
    const float* fc_b     = (const float*)d_in[15];
    const float* out_w    = (const float*)d_in[16];
    const float* out_b    = (const float*)d_in[17];

    char* ws = (char*)d_ws;
    size_t off = 0;
    float* x    = (float*)(ws + off); off += (size_t)NATOMS * AF * 4;
    bf16* x_bf  = (bf16*)(ws + off);  off += (size_t)NATOMS * AF * 2;
    bf16* nb_bf = (bf16*)(ws + off);  off += (size_t)NATOMS * MNBR * NBRF * 2;
    bf16* w1t   = (bf16*)(ws + off);  off += (size_t)NCONV * N1 * KDIM * 2;
    bf16* w2t   = (bf16*)(ws + off);  off += (size_t)NCONV * AF * N1 * 2;
    bf16* ns_bf = (bf16*)(ws + off);  off += (size_t)NATOMS * N1 * 2;
    float* y    = (float*)(ws + off); off += (size_t)NATOMS * AF * 4;
    float* bnacc = (float*)(ws + off); off += (size_t)NCONV * 512 * 4;
    (void)ws_size; (void)in_sizes; (void)n_in; (void)out_size;

    prep_weights<<<4992, 256, 0, stream>>>(w1, w2, w1t, w2t, bnacc);
    prep_nbr<<<(NATOMS * MNBR * NBRF + 255) / 256, 256, 0, stream>>>(
        nbr_fea, nb_bf, NATOMS * MNBR * NBRF);
    embed_k<<<NATOMS, 256, 0, stream>>>(atom_fea, emb_w, emb_b, x, x_bf);

    for (int l = 0; l < NCONV; ++l) {
        conv_gemm<<<dim3(NATOMS / APB, 2), 256, 0, stream>>>(
            x_bf, nb_bf, nbr_idx, w1t, b1, ns_bf, l);
        gemm2_stats<<<(NATOMS + MT2 - 1) / MT2, 256, 0, stream>>>(
            ns_bf, w2t, b2, y, bnacc, l);
        bn_apply<<<NATOMS, 256, 0, stream>>>(x, y, bnacc, gamma, beta, x_bf, l);
    }

    head_k<<<BCRYS, 256, 0, stream>>>(x, m1_index, m2_fea, fc_w, fc_b,
                                      out_w, out_b, (float*)d_out);
}
